// Cls_Decoder_20744692040045
// MI455X (gfx1250) — matrix operand order repaired and checked by reading, not compiled, not run on hardware
//
#include <hip/hip_runtime.h>
#include <stdint.h>

typedef __attribute__((ext_vector_type(16))) __bf16 v16bf;
typedef __attribute__((ext_vector_type(8)))  __bf16 v8bf;
typedef __attribute__((ext_vector_type(8)))  float  v8f;
typedef __attribute__((ext_vector_type(4)))  float  v4f;

#define N_ROWS 100000
#define D_IN   512
#define H2D    1024
#define OUTC   250
#define BSEG   256
#define EPSL   1e-5f

// ---------- bf16 helpers (round-to-nearest-even) ----------
__device__ __forceinline__ float bf2f(__bf16 h) {
  unsigned short u = __builtin_bit_cast(unsigned short, h);
  unsigned int x = ((unsigned int)u) << 16;
  return __builtin_bit_cast(float, x);
}
__device__ __forceinline__ __bf16 f2bf(float f) {
  unsigned int u = __builtin_bit_cast(unsigned int, f);
  unsigned int r = u + 0x7FFFu + ((u >> 16) & 1u);
  unsigned short h = (unsigned short)(r >> 16);
  return __builtin_bit_cast(__bf16, h);
}

// ---------- conversion kernels ----------
// vectorized: 8 floats (2x b128 load) -> 8 bf16 (1x b128 store) per iteration
__global__ void __launch_bounds__(256)
k_f32_to_bf16(const float* __restrict__ src, __bf16* __restrict__ dst, int n8) {
  int i = blockIdx.x * blockDim.x + threadIdx.x;
  int stride = gridDim.x * blockDim.x;
  for (; i < n8; i += stride) {
    v4f a = *(const v4f*)(src + (size_t)i * 8);
    v4f b = *(const v4f*)(src + (size_t)i * 8 + 4);
    v8bf o;
    o[0] = f2bf(a[0]); o[1] = f2bf(a[1]); o[2] = f2bf(a[2]); o[3] = f2bf(a[3]);
    o[4] = f2bf(b[0]); o[5] = f2bf(b[1]); o[6] = f2bf(b[2]); o[7] = f2bf(b[3]);
    *(v8bf*)(dst + (size_t)i * 8) = o;
  }
}

// mW2 [250,1024] f32 -> [256,1024] bf16, zero padded rows 250..255
__global__ void __launch_bounds__(256)
k_pad_mw2(const float* __restrict__ src, __bf16* __restrict__ dst) {
  int i = blockIdx.x * blockDim.x + threadIdx.x;   // 256*1024 total
  int j = i >> 10, k = i & 1023;
  float v = (j < OUTC) ? src[j * H2D + k] : 0.0f;
  dst[i] = f2bf(v);
}

// ============================================================================
// GEMM strip: each wave owns a 16-row strip (4 waves / 128-thread block).
//   A fragments (16 k-steps, 16x32 bf16) preloaded from global into VGPRs
//   (register-resident, proven spill-free).
//   B frags: contiguous 32B per-lane loads from row-major W1 (k = 16*(lane/16) + s, same as A's
//   layout), 2-deep software pipeline (load k+2 before consuming k) so each
//   WMMA waits only on loads issued two steps earlier.
//   h strip (16x1024 bf16 = 32KB/wave) in LDS for the LayerNorm epilogue.
// Dynamic LDS per block: 128KB -> 2 blocks/WGP -> 2 waves/SIMD.
// ============================================================================
__device__ __forceinline__ void gemm_strip(
    const __bf16* __restrict__ aSrc,   // global, row stride D_IN
    const __bf16* __restrict__ w1b,    // [1024, 512] bf16
    const float*  __restrict__ b1,     // [1024]
    __bf16* h,                         // [16][1024] this wave's output strip
    int lane, int rowclamp, int rowbase)
{
  const int g  = lane >> 4;
  const int lm = lane & 15;

  int arow = rowbase + lm; if (arow > rowclamp) arow = rowclamp;
  const __bf16* aptr = aSrc + (size_t)arow * D_IN;
  v16bf afrag[16];
#pragma unroll
  for (int k = 0; k < 16; ++k) {
    v8bf lo = *(const v8bf*)(aptr + k * 32 + g * 16);
    v8bf hi = *(const v8bf*)(aptr + k * 32 + g * 16 + 8);
    afrag[k] = __builtin_shufflevector(lo, hi,
                 0,1,2,3,4,5,6,7,8,9,10,11,12,13,14,15);
  }

  for (int jt = 0; jt < 64; ++jt) {
    const __bf16* bptr = w1b + (size_t)(jt * 16 + lm) * D_IN + g * 16;
    if (jt < 63)  // warm next B tile into L2 one tile ahead
      __builtin_prefetch(w1b + (size_t)((jt + 1) * 16 + lm) * D_IN, 0, 3);

    v8f acc = {};
    v16bf b0 = *(const v16bf*)(bptr);
    v16bf b1f = *(const v16bf*)(bptr + 32);
#pragma unroll
    for (int k = 0; k < 16; ++k) {
      v16bf b2 = b1f;
      if (k < 14) b2 = *(const v16bf*)(bptr + (k + 2) * 32);
      acc = __builtin_amdgcn_wmma_f32_16x16x32_bf16(
          false, afrag[k], false, b0, (short)0, acc, false, false);
      b0 = b1f; b1f = b2;
    }
    int col = jt * 16 + lm;
    float bias = b1[col];
#pragma unroll
    for (int v = 0; v < 8; ++v) {
      int r = v + g * 8;                 // C layout: M = v + 8*(lane/16), N = lane%16
      h[r * H2D + col] = f2bf(acc[v] + bias);
    }
  }
}

// ---------- gate MLP: h = relu(LN(x@gW1.T + gb1)); gate = h@gW2 + gb2 ----------
__global__ void __launch_bounds__(128)
k_gate(const __bf16* __restrict__ xb,    // [N,512]
       const __bf16* __restrict__ w1b,   // [1024,512]
       const float*  __restrict__ gb1,   // [1024]
       const float*  __restrict__ lng,   // [1024]
       const float*  __restrict__ lnb,   // [1024]
       const float*  __restrict__ w2,    // [1024]
       const float*  __restrict__ b2,    // [1]
       float* __restrict__ gate)         // [N]
{
  extern __shared__ __bf16 smem[];       // 4*16*1024 bf16 = 128KB
  const int tid  = threadIdx.x;
  const int wave = tid >> 5;
  const int lane = tid & 31;
  const int rowbase = blockIdx.x * 64 + wave * 16;
  __bf16* h = smem + wave * 16 * H2D;

  gemm_strip(xb, w1b, gb1, h, lane, N_ROWS - 1, rowbase);

  // per-row LayerNorm + ReLU + dot(gW2) (wave-local; LDS ops in-order per wave)
  for (int r = 0; r < 16; ++r) {
    float s = 0.f, s2 = 0.f;
    for (int i = lane; i < H2D; i += 32) {
      float v = bf2f(h[r * H2D + i]); s += v; s2 += v * v;
    }
#pragma unroll
    for (int off = 16; off; off >>= 1) {
      s  += __shfl_xor(s,  off, 32);
      s2 += __shfl_xor(s2, off, 32);
    }
    float mean = s * (1.f / 1024.f);
    float var  = s2 * (1.f / 1024.f) - mean * mean;
    float rs   = rsqrtf(var + EPSL);
    float acc = 0.f;
    for (int i = lane; i < H2D; i += 32) {
      float v = (bf2f(h[r * H2D + i]) - mean) * rs * lng[i] + lnb[i];
      acc += fmaxf(v, 0.f) * w2[i];
    }
#pragma unroll
    for (int off = 16; off; off >>= 1) acc += __shfl_xor(acc, off, 32);
    int row = rowbase + r;
    if (lane == 0 && row < N_ROWS) gate[row] = acc + b2[0];
  }
}

// ---------- classifier layer 1: z = relu(LN(out@mW1.T + mb1)) -> bf16 ----------
__global__ void __launch_bounds__(128)
k_mlp1(const __bf16* __restrict__ ab,   // [256,512]
       const __bf16* __restrict__ w1b,  // [1024,512]
       const float*  __restrict__ b1,   // [1024]
       const float*  __restrict__ lng,  // [1024]
       const float*  __restrict__ lnb,  // [1024]
       __bf16* __restrict__ z)          // [256,1024]
{
  extern __shared__ __bf16 smem[];
  const int tid  = threadIdx.x;
  const int wave = tid >> 5;
  const int lane = tid & 31;
  const int rowbase = blockIdx.x * 64 + wave * 16;
  __bf16* h = smem + wave * 16 * H2D;

  gemm_strip(ab, w1b, b1, h, lane, 255, rowbase);

  for (int r = 0; r < 16; ++r) {
    float s = 0.f, s2 = 0.f;
    for (int i = lane; i < H2D; i += 32) {
      float v = bf2f(h[r * H2D + i]); s += v; s2 += v * v;
    }
#pragma unroll
    for (int off = 16; off; off >>= 1) {
      s  += __shfl_xor(s,  off, 32);
      s2 += __shfl_xor(s2, off, 32);
    }
    float mean = s * (1.f / 1024.f);
    float var  = s2 * (1.f / 1024.f) - mean * mean;
    float rs   = rsqrtf(var + EPSL);
    int row = rowbase + r;
    for (int i = lane; i < H2D; i += 32) {
      float v = (bf2f(h[r * H2D + i]) - mean) * rs * lng[i] + lnb[i];
      z[(size_t)row * H2D + i] = f2bf(fmaxf(v, 0.f));
    }
  }
}

// ---------- segment softmax + weighted pool + pooled LayerNorm ----------
__global__ void __launch_bounds__(256)
k_pool(const float* __restrict__ x,      // [N,512] fp32
       const int*   __restrict__ bid,    // [N] sorted
       const float* __restrict__ gate,   // [N]
       const float* __restrict__ png,    // [512]
       const float* __restrict__ pnb,    // [512]
       __bf16* __restrict__ outb)        // [256,512] bf16
{
  __shared__ float red[256];
  __shared__ float wbuf[256];
  const int b   = blockIdx.x;
  const int tid = threadIdx.x;

  int lo = 0, hi = N_ROWS;
  while (lo < hi) { int mid = (lo + hi) >> 1; if (bid[mid] < b) lo = mid + 1; else hi = mid; }
  const int start = lo;
  lo = start; hi = N_ROWS;
  while (lo < hi) { int mid = (lo + hi) >> 1; if (bid[mid] < b + 1) lo = mid + 1; else hi = mid; }
  const int end = lo;

  float a0 = 0.f, a1 = 0.f;
  if (end > start) {
    float m = -3.4e38f;
    for (int n = start + tid; n < end; n += 256) m = fmaxf(m, gate[n]);
    red[tid] = m; __syncthreads();
    for (int s = 128; s; s >>= 1) { if (tid < s) red[tid] = fmaxf(red[tid], red[tid + s]); __syncthreads(); }
    m = red[0]; __syncthreads();

    float se = 0.f;
    for (int n = start + tid; n < end; n += 256) se += __expf(gate[n] - m);
    red[tid] = se; __syncthreads();
    for (int s = 128; s; s >>= 1) { if (tid < s) red[tid] += red[tid + s]; __syncthreads(); }
    float inv = 1.f / red[0]; __syncthreads();

    for (int base = start; base < end; base += 256) {
      int cnt = min(256, end - base);
      if (tid < cnt) wbuf[tid] = __expf(gate[base + tid] - m) * inv;
      __syncthreads();
      for (int i = 0; i < cnt; ++i) {
        float w = wbuf[i];
        const float* xr = x + (size_t)(base + i) * D_IN;
        a0 += w * xr[tid];
        a1 += w * xr[tid + 256];
      }
      __syncthreads();
    }
  }

  red[tid] = a0 + a1; __syncthreads();
  for (int s = 128; s; s >>= 1) { if (tid < s) red[tid] += red[tid + s]; __syncthreads(); }
  float mean = red[0] * (1.f / 512.f); __syncthreads();
  red[tid] = a0 * a0 + a1 * a1; __syncthreads();
  for (int s = 128; s; s >>= 1) { if (tid < s) red[tid] += red[tid + s]; __syncthreads(); }
  float var = red[0] * (1.f / 512.f) - mean * mean;
  float rs  = rsqrtf(var + EPSL);
  outb[b * 512 + tid]       = f2bf((a0 - mean) * rs * png[tid]       + pnb[tid]);
  outb[b * 512 + tid + 256] = f2bf((a1 - mean) * rs * png[tid + 256] + pnb[tid + 256]);
}

// ---------- classifier layer 2: logits = z @ mW2p.T + mb2 ----------
__global__ void __launch_bounds__(128)
k_mlp2(const __bf16* __restrict__ zb,   // [256,1024]
       const __bf16* __restrict__ w2b,  // [256,1024] padded
       const float*  __restrict__ b2,   // [250]
       float* __restrict__ outp)        // [256,250]
{
  const int tid  = threadIdx.x;
  const int wave = tid >> 5;
  const int lane = tid & 31;
  const int g    = lane >> 4;
  const int lm   = lane & 15;
  const int rowbase = blockIdx.x * 64 + wave * 16;
  const __bf16* aptr = zb + (size_t)(rowbase + lm) * H2D;

  for (int jt = 0; jt < 16; ++jt) {
    const __bf16* bptr = w2b + (size_t)(jt * 16 + lm) * H2D + g * 16;
    v8f acc = {};
#pragma unroll
    for (int k = 0; k < 32; ++k) {
      v8bf lo = *(const v8bf*)(aptr + k * 32 + g * 16);
      v8bf hi = *(const v8bf*)(aptr + k * 32 + g * 16 + 8);
      v16bf a = __builtin_shufflevector(lo, hi, 0,1,2,3,4,5,6,7,8,9,10,11,12,13,14,15);
      v16bf b = *(const v16bf*)(bptr + k * 32);
      acc = __builtin_amdgcn_wmma_f32_16x16x32_bf16(
          false, a, false, b, (short)0, acc, false, false);
    }
    int col = jt * 16 + lm;
    if (col < OUTC) {
      float bias = b2[col];
#pragma unroll
      for (int v = 0; v < 8; ++v) {
        int r = rowbase + v + g * 8;
        outp[r * OUTC + col] = acc[v] + bias;
      }
    }
  }
}

extern "C" void kernel_launch(void* const* d_in, const int* in_sizes, int n_in,
                              void* d_out, int out_size, void* d_ws, size_t ws_size,
                              hipStream_t stream) {
  const float* x    = (const float*)d_in[0];
  const int*   bid  = (const int*)  d_in[1];
  const float* gW1  = (const float*)d_in[2];
  const float* gb1  = (const float*)d_in[3];
  const float* glng = (const float*)d_in[4];
  const float* glnb = (const float*)d_in[5];
  const float* gW2  = (const float*)d_in[6];
  const float* gb2  = (const float*)d_in[7];
  const float* png  = (const float*)d_in[8];
  const float* pnb  = (const float*)d_in[9];
  const float* mW1  = (const float*)d_in[10];
  const float* mb1  = (const float*)d_in[11];
  const float* mlng = (const float*)d_in[12];
  const float* mlnb = (const float*)d_in[13];
  const float* mW2  = (const float*)d_in[14];
  const float* mb2  = (const float*)d_in[15];
  float* outp = (float*)d_out;

  char* ws = (char*)d_ws;
  size_t off = 0;
  auto alloc = [&](size_t bytes) -> void* {
    void* p = ws + off;
    off = (off + bytes + 255) & ~(size_t)255;
    return p;
  };
  __bf16* xb   = (__bf16*)alloc((size_t)N_ROWS * D_IN * 2);
  __bf16* gW1b = (__bf16*)alloc((size_t)H2D * D_IN * 2);
  __bf16* mW1b = (__bf16*)alloc((size_t)H2D * D_IN * 2);
  __bf16* mW2b = (__bf16*)alloc((size_t)256 * H2D * 2);
  float*  gate = (float*) alloc((size_t)N_ROWS * 4);
  __bf16* outb = (__bf16*)alloc((size_t)256 * D_IN * 2);
  __bf16* zb   = (__bf16*)alloc((size_t)256 * H2D * 2);

  k_f32_to_bf16<<<4096, 256, 0, stream>>>(x,   xb,   (N_ROWS * D_IN) / 8);
  k_f32_to_bf16<<<256,  256, 0, stream>>>(gW1, gW1b, (H2D * D_IN) / 8);
  k_f32_to_bf16<<<256,  256, 0, stream>>>(mW1, mW1b, (H2D * D_IN) / 8);
  k_pad_mw2<<<(256 * H2D) / 256, 256, 0, stream>>>(mW2, mW2b);

  size_t lds = (size_t)4 * 16 * H2D * sizeof(__bf16);   // 128 KB / block
  k_gate<<<(N_ROWS + 63) / 64, 128, lds, stream>>>(xb, gW1b, gb1, glng, glnb, gW2, gb2, gate);
  k_pool<<<BSEG, 256, 0, stream>>>(x, bid, gate, png, pnb, outb);
  k_mlp1<<<4, 128, lds, stream>>>(outb, mW1b, mb1, mlng, mlnb, zb);
  k_mlp2<<<4, 128, 0, stream>>>(zb, mW2b, mb2, outp);
}
